// EquivariantMLP_65618510349072
// MI455X (gfx1250) — compile-verified, matrix operand order checked by reading, not run on hardware
//
#include <hip/hip_runtime.h>
#include <hip/hip_bf16.h>

// ---------------------------------------------------------------------------
// Problem constants (from reference)
// ---------------------------------------------------------------------------
#define N_NODES 20000
#define E_EDGES 640000
#define DS 128
#define DV 64
#define LN_EPS 1e-5f

#define NWAVES 8            // waves per block (256 threads, wave32)

typedef __attribute__((ext_vector_type(16))) _Float16 h16;
typedef __attribute__((ext_vector_type(8)))  _Float16 h8;
typedef __attribute__((ext_vector_type(4)))  _Float16 h4;
typedef __attribute__((ext_vector_type(8)))  float    f8;
typedef __attribute__((ext_vector_type(4)))  float    f4;

__device__ __forceinline__ float silu_f(float x) {
    return x / (1.0f + __expf(-x));
}

// ---------------------------------------------------------------------------
// Convert a KxN row-major f32 weight matrix into WMMA B-fragment-swizzled
// f16 layout in LDS. Fragment (kt,nt) = 512 halves; within a fragment,
// half index = lane*16 + h, holding W[kt*32 + koff][nt*16 + lane%16] with
//   koff = (h<8 ? h : h+8) + (lane>=16 ? 8 : 0)
// matching the CDNA5 16-bit 32x16 B-matrix VGPR layout, so the consumer
// does ONE contiguous 32-byte LDS read per lane per fragment.
// ---------------------------------------------------------------------------
__device__ __forceinline__ void fill_weight_frags(const float* __restrict__ Wg,
                                                  _Float16* __restrict__ dst,
                                                  int K, int N, int tid, int nthreads) {
    const int total = K * N;
    const int ntiles_n = N >> 4;
    for (int idx = tid; idx < total; idx += nthreads) {
        int frag   = idx >> 9;          // /512
        int within = idx & 511;
        int lane   = within >> 4;
        int h      = within & 15;
        int kt     = frag / ntiles_n;
        int nt     = frag - kt * ntiles_n;
        int k      = kt * 32 + ((h < 8) ? h : (h + 8)) + ((lane >= 16) ? 8 : 0);
        int n      = nt * 16 + (lane & 15);
        dst[idx] = (_Float16)Wg[(size_t)k * N + n];
    }
}

// B fragment: one 32B contiguous LDS read per lane.
__device__ __forceinline__ h16 load_b_frag(const _Float16* __restrict__ Wlds,
                                           int frag, int lane) {
    return *(const h16*)(Wlds + ((size_t)frag << 9) + (lane << 4));
}

// A fragment from a row-major [16 x stride] f16 LDS tile.
// Lane layout (CDNA5 16-bit 16x32 A): M = lane%16; lanes>=16 take K offset +8;
// halves 0..7 -> K = kbase+sel+0..7, halves 8..15 -> K = kbase+16+sel+0..7.
__device__ __forceinline__ h16 load_a_frag(const _Float16* __restrict__ T,
                                           int laneM, int selHi8, int kt, int stride) {
    const _Float16* ap = T + laneM * stride + kt * 32 + selHi8;
    h8 lo = *(const h8*)(ap);
    h8 hi = *(const h8*)(ap + 16);
    return __builtin_shufflevector(lo, hi, 0,1,2,3,4,5,6,7,8,9,10,11,12,13,14,15);
}

__device__ __forceinline__ f8 splat8(float x) {
    f8 c = {x, x, x, x, x, x, x, x};
    return c;
}

// ---------------------------------------------------------------------------
// LDS budgets
// ---------------------------------------------------------------------------
#define EDGE_W_BYTES   (98304 + 32768 + 16384)        // W1 + W2 + Wv as f16 frags
#define EDGE_PW_BYTES  20736                          // Atile+Ttile+Mtile+rowI+evT
#define EDGE_SMEM      (EDGE_W_BYTES + NWAVES * EDGE_PW_BYTES)   // 313344 B

#define NODE_W_BYTES   32768
#define NODE_PW_BYTES  12288                          // Atile f16 + Smid f32
#define NODE_SMEM      (NODE_W_BYTES + NWAVES * NODE_PW_BYTES)   // 131072 B

// ---------------------------------------------------------------------------
// Kernel 0: zero the scatter accumulators in d_ws
// ---------------------------------------------------------------------------
__global__ void zero_ws_kernel(float* __restrict__ ws, int n) {
    int i = blockIdx.x * blockDim.x + threadIdx.x;
    if (i < n) ws[i] = 0.0f;
}

// ---------------------------------------------------------------------------
// Kernel 1: fused edge MLP + scatter.
// Each wave processes a 16-edge tile entirely through WMMA.
// GEMM loops: kt outer, all N-tile accumulators live in registers, so each
// A fragment is loaded from LDS exactly once per k-step.
// ---------------------------------------------------------------------------
__global__ void __launch_bounds__(NWAVES * 32, 1)
edge_kernel(const float* __restrict__ s,
            const int*   __restrict__ eidx,
            const float* __restrict__ eattr,
            const float* __restrict__ evu,
            const float* __restrict__ W1, const float* __restrict__ b1,
            const float* __restrict__ W2, const float* __restrict__ b2,
            const float* __restrict__ Wv, const float* __restrict__ bv,
            float* __restrict__ sAcc,      // [N_NODES][128]
            float* __restrict__ vAcc)      // [N_NODES][64][3]
{
    extern __shared__ char smem[];
    _Float16* W1s = (_Float16*)smem;                  // 384x128 frags
    _Float16* W2s = W1s + 384 * 128;                  // 128x128 frags
    _Float16* Wvs = W2s + 128 * 128;                  // 128x64  frags
    char* pw_base = smem + EDGE_W_BYTES;

    const int tid  = threadIdx.x;
    const int lane = tid & 31;
    const int wave = tid >> 5;
    const int laneM  = lane & 15;
    const int selHi8 = (lane >> 4) << 3;  // 0 / 8

    char* my = pw_base + wave * EDGE_PW_BYTES;
    _Float16* Atile = (_Float16*)my;                  // [16][384]
    _Float16* Ttile = Atile + 16 * 384;               // [16][128]
    _Float16* Mtile = Ttile + 16 * 128;               // [16][128]
    int*      rowI  = (int*)(Mtile + 16 * 128);       // [16]
    float*    evT   = (float*)(rowI + 16);            // [16][3]

    // One-time per-block: stage all weights as f16 B-fragments in LDS.
    fill_weight_frags(W1, W1s, 3 * DS, DS, tid, blockDim.x);
    fill_weight_frags(W2, W2s, DS, DS, tid, blockDim.x);
    fill_weight_frags(Wv, Wvs, DS, DV, tid, blockDim.x);
    __syncthreads();

    const int nTiles = E_EDGES / 16;   // 40000
    for (int tile = blockIdx.x * NWAVES + wave; tile < nTiles;
         tile += gridDim.x * NWAVES) {
        const int eBase = tile * 16;

        // Row indices + unit edge vectors for this tile.
        if (lane < 16) {
            int e = eBase + lane;
            rowI[lane]        = eidx[e];
            evT[lane * 3 + 0] = evu[(size_t)e * 3 + 0];
            evT[lane * 3 + 1] = evu[(size_t)e * 3 + 1];
            evT[lane * 3 + 2] = evu[(size_t)e * 3 + 2];
        }

        // Stage A tile: [s_row | s_col | edge_attr] per edge, as f16.
        for (int m = 0; m < 16; ++m) {
            int e = eBase + m;
            int r = eidx[e];             // uniform within wave -> scalar load
            int c = eidx[E_EDGES + e];
            f4 fr = *(const f4*)(s     + (size_t)r * DS + lane * 4);
            f4 fc = *(const f4*)(s     + (size_t)c * DS + lane * 4);
            f4 fa = *(const f4*)(eattr + (size_t)e * DS + lane * 4);
            h4 hr, hc, ha;
            #pragma unroll
            for (int j = 0; j < 4; ++j) {
                hr[j] = (_Float16)fr[j];
                hc[j] = (_Float16)fc[j];
                ha[j] = (_Float16)fa[j];
            }
            *(h4*)(Atile + m * 384 +       lane * 4) = hr;
            *(h4*)(Atile + m * 384 + 128 + lane * 4) = hc;
            *(h4*)(Atile + m * 384 + 256 + lane * 4) = ha;
        }

        // ---- Layer 1: t = silu(A[16x384] @ W1[384x128] + b1) ----
        {
            f8 acc[8];
            #pragma unroll
            for (int nt = 0; nt < 8; ++nt) acc[nt] = splat8(b1[nt * 16 + laneM]);
            for (int kt = 0; kt < 12; ++kt) {
                h16 a = load_a_frag(Atile, laneM, selHi8, kt, 384);
                #pragma unroll
                for (int nt = 0; nt < 8; ++nt) {
                    h16 b = load_b_frag(W1s, kt * 8 + nt, lane);
                    acc[nt] = __builtin_amdgcn_wmma_f32_16x16x32_f16(
                                  false, a, false, b, (short)0, acc[nt], false, false);
                }
            }
            #pragma unroll
            for (int nt = 0; nt < 8; ++nt) {
                #pragma unroll
                for (int r = 0; r < 8; ++r) {
                    int M = r + selHi8;
                    Ttile[M * 128 + nt * 16 + laneM] = (_Float16)silu_f(acc[nt][r]);
                }
            }
        }

        // ---- Layer 2: msg = t @ W2 + b2; scatter msg; keep silu(msg) ----
        {
            f8 acc[8];
            #pragma unroll
            for (int nt = 0; nt < 8; ++nt) acc[nt] = splat8(b2[nt * 16 + laneM]);
            for (int kt = 0; kt < 4; ++kt) {
                h16 a = load_a_frag(Ttile, laneM, selHi8, kt, 128);
                #pragma unroll
                for (int nt = 0; nt < 8; ++nt) {
                    h16 b = load_b_frag(W2s, kt * 8 + nt, lane);
                    acc[nt] = __builtin_amdgcn_wmma_f32_16x16x32_f16(
                                  false, a, false, b, (short)0, acc[nt], false, false);
                }
            }
            #pragma unroll
            for (int nt = 0; nt < 8; ++nt) {
                #pragma unroll
                for (int r = 0; r < 8; ++r) {
                    int M = r + selHi8;
                    float msg = acc[nt][r];
                    atomicAdd(&sAcc[(size_t)rowI[M] * DS + nt * 16 + laneM], msg);
                    Mtile[M * 128 + nt * 16 + laneM] = (_Float16)silu_f(msg);
                }
            }
        }

        // ---- Vector head: g = silu(msg) @ Wv + bv; scatter g * ev_unit ----
        {
            f8 acc[4];
            #pragma unroll
            for (int nt = 0; nt < 4; ++nt) acc[nt] = splat8(bv[nt * 16 + laneM]);
            for (int kt = 0; kt < 4; ++kt) {
                h16 a = load_a_frag(Mtile, laneM, selHi8, kt, 128);
                #pragma unroll
                for (int nt = 0; nt < 4; ++nt) {
                    h16 b = load_b_frag(Wvs, kt * 4 + nt, lane);
                    acc[nt] = __builtin_amdgcn_wmma_f32_16x16x32_f16(
                                  false, a, false, b, (short)0, acc[nt], false, false);
                }
            }
            #pragma unroll
            for (int nt = 0; nt < 4; ++nt) {
                #pragma unroll
                for (int r = 0; r < 8; ++r) {
                    int M   = r + selHi8;
                    float g = acc[nt][r];
                    int row = rowI[M];
                    float e0 = evT[M * 3 + 0];
                    float e1 = evT[M * 3 + 1];
                    float e2 = evT[M * 3 + 2];
                    size_t base = ((size_t)row * DV + nt * 16 + laneM) * 3;
                    atomicAdd(&vAcc[base + 0], g * e0);
                    atomicAdd(&vAcc[base + 1], g * e1);
                    atomicAdd(&vAcc[base + 2], g * e2);
                }
            }
        }
    }
}

// ---------------------------------------------------------------------------
// Kernel 2: node update  s_new = LN(s + silu(sAcc) @ Ws + bs)
// ---------------------------------------------------------------------------
__global__ void __launch_bounds__(NWAVES * 32, 2)
node_kernel(const float* __restrict__ s,
            const float* __restrict__ Ws, const float* __restrict__ bs,
            const float* __restrict__ gamma, const float* __restrict__ beta,
            const float* __restrict__ sAcc,
            float* __restrict__ outS)
{
    extern __shared__ char smem[];
    _Float16* Wss = (_Float16*)smem;                  // 128x128 frags
    char* pw_base = smem + NODE_W_BYTES;

    const int tid  = threadIdx.x;
    const int lane = tid & 31;
    const int wave = tid >> 5;
    const int laneM  = lane & 15;
    const int selHi8 = (lane >> 4) << 3;

    char* my = pw_base + wave * NODE_PW_BYTES;
    _Float16* Atile = (_Float16*)my;                  // [16][128] f16
    float*    Smid  = (float*)(Atile + 16 * 128);     // [16][128] f32

    fill_weight_frags(Ws, Wss, DS, DS, tid, blockDim.x);
    __syncthreads();

    const int nTiles = N_NODES / 16;   // 1250
    for (int tile = blockIdx.x * NWAVES + wave; tile < nTiles;
         tile += gridDim.x * NWAVES) {
        const int nodeBase = tile * 16;

        // Stage A = silu(sAcc) as f16.
        for (int m = 0; m < 16; ++m) {
            int node = nodeBase + m;
            f4 f = *(const f4*)(sAcc + (size_t)node * DS + lane * 4);
            h4 h;
            #pragma unroll
            for (int j = 0; j < 4; ++j) h[j] = (_Float16)silu_f(f[j]);
            *(h4*)(Atile + m * 128 + lane * 4) = h;
        }

        // A @ Ws + bs -> Smid (f32 in LDS)
        {
            f8 acc[8];
            #pragma unroll
            for (int nt = 0; nt < 8; ++nt) acc[nt] = splat8(bs[nt * 16 + laneM]);
            for (int kt = 0; kt < 4; ++kt) {
                h16 a = load_a_frag(Atile, laneM, selHi8, kt, 128);
                #pragma unroll
                for (int nt = 0; nt < 8; ++nt) {
                    h16 b = load_b_frag(Wss, kt * 8 + nt, lane);
                    acc[nt] = __builtin_amdgcn_wmma_f32_16x16x32_f16(
                                  false, a, false, b, (short)0, acc[nt], false, false);
                }
            }
            #pragma unroll
            for (int nt = 0; nt < 8; ++nt) {
                #pragma unroll
                for (int r = 0; r < 8; ++r)
                    Smid[(r + selHi8) * 128 + nt * 16 + laneM] = acc[nt][r];
            }
        }

        // Residual + LayerNorm: lanes 0..15 each own one row.
        if (lane < 16) {
            int m = lane, node = nodeBase + m;
            float sum = 0.0f, sq = 0.0f;
            for (int k = 0; k < DS; ++k) {
                float x = Smid[m * 128 + k] + s[(size_t)node * DS + k];
                Smid[m * 128 + k] = x;
                sum += x;
                sq  += x * x;
            }
            float mu   = sum * (1.0f / DS);
            float var  = sq * (1.0f / DS) - mu * mu;
            float rstd = rsqrtf(var + LN_EPS);
            for (int k = 0; k < DS; ++k) {
                outS[(size_t)node * DS + k] =
                    gamma[k] * (Smid[m * 128 + k] - mu) * rstd + beta[k];
            }
        }
    }
}

// ---------------------------------------------------------------------------
// Kernel 3: v_new = v + vAcc (elementwise)
// ---------------------------------------------------------------------------
__global__ void vadd_kernel(const float* __restrict__ v,
                            const float* __restrict__ vAcc,
                            float* __restrict__ outV, int n) {
    int i = blockIdx.x * blockDim.x + threadIdx.x;
    if (i < n) outV[i] = v[i] + vAcc[i];
}

// ---------------------------------------------------------------------------
// Host launcher
// ---------------------------------------------------------------------------
extern "C" void kernel_launch(void* const* d_in, const int* in_sizes, int n_in,
                              void* d_out, int out_size, void* d_ws, size_t ws_size,
                              hipStream_t stream) {
    (void)in_sizes; (void)n_in; (void)out_size; (void)ws_size;

    const float* s     = (const float*)d_in[0];
    const float* v     = (const float*)d_in[1];
    const int*   eidx  = (const int*)  d_in[2];
    const float* eattr = (const float*)d_in[3];
    const float* evu   = (const float*)d_in[4];
    const float* W1    = (const float*)d_in[5];
    const float* b1    = (const float*)d_in[6];
    const float* W2    = (const float*)d_in[7];
    const float* b2    = (const float*)d_in[8];
    const float* Wsm   = (const float*)d_in[9];
    const float* bsb   = (const float*)d_in[10];
    const float* Wv    = (const float*)d_in[11];
    const float* bv    = (const float*)d_in[12];
    const float* gamma = (const float*)d_in[13];
    const float* beta  = (const float*)d_in[14];

    float* outS = (float*)d_out;
    float* outV = outS + (size_t)N_NODES * DS;

    float* sAcc = (float*)d_ws;                       // N*128 f32
    float* vAcc = sAcc + (size_t)N_NODES * DS;        // N*64*3 f32

    // Allow >64KB dynamic LDS (idempotent, capture-safe host call).
    hipFuncSetAttribute((const void*)edge_kernel,
                        hipFuncAttributeMaxDynamicSharedMemorySize, EDGE_SMEM);
    hipFuncSetAttribute((const void*)node_kernel,
                        hipFuncAttributeMaxDynamicSharedMemorySize, NODE_SMEM);

    // 0) zero accumulators (must run every call: atomics accumulate)
    {
        int n = N_NODES * (DS + DV * 3);              // 6.4M floats
        zero_ws_kernel<<<(n + 255) / 256, 256, 0, stream>>>(sAcc, n);
    }
    // 1) fused edge MLP + scatter
    edge_kernel<<<640, NWAVES * 32, EDGE_SMEM, stream>>>(
        s, eidx, eattr, evu, W1, b1, W2, b2, Wv, bv, sAcc, vAcc);
    // 2) node update + LayerNorm
    node_kernel<<<(N_NODES / 16 + NWAVES - 1) / NWAVES, NWAVES * 32, NODE_SMEM, stream>>>(
        s, Wsm, bsb, gamma, beta, sAcc, outS);
    // 3) vector residual
    {
        int n = N_NODES * DV * 3;
        vadd_kernel<<<(n + 255) / 256, 256, 0, stream>>>(v, vAcc, outV, n);
    }
}
